// ReadoutResNet_19576460935856
// MI455X (gfx1250) — matrix operand order repaired and checked by reading, not compiled, not run on hardware
//
#include <hip/hip_runtime.h>
#include <hip/hip_bf16.h>
#include <math.h>

typedef __attribute__((ext_vector_type(16))) _Float16 v16h;
typedef __attribute__((ext_vector_type(8)))  _Float16 v8h;
typedef __attribute__((ext_vector_type(8)))  float    v8f;
typedef __attribute__((ext_vector_type(4)))  float    f4v;

#define BATCH   4096
#define SEQ     120
#define DIM     256
#define GRID_HW 12
#define RDIM    48
#define NLAYER  16
#define NPOS    36
#define BPW     4      // batches per workgroup
#define NCOLT   9      // column tiles of 16 per workgroup (BPW*NPOS/16)

#define PACKA_ELEMS (16*32*32*16)   // 16 M-tiles x 32 K-tiles x 32 lanes x 16 halfs
#define PACKR_ELEMS (3*8*32*16)     // 3 M-tiles x 8 K-tiles

// Branch-free GELU (tanh form as x * sigmoid(2*0.7978845608*(x+0.044715x^3))):
// ~7 VALU ops incl. v_exp_f32 + v_rcp_f32; saturates correctly at +/-inf args.
__device__ __forceinline__ float gelu_fast(float x) {
    float q   = __builtin_fmaf(0.044715f, x * x, 1.0f);
    float arg = x * q * -2.3022159f;               // -2*0.7978845608*log2(e)
    float e   = __builtin_amdgcn_exp2f(arg);
    return x * __builtin_amdgcn_rcpf(1.0f + e);
}

// ---------------------------------------------------------------------------
// Pack conv_w (256,256,2,2) and red_w (48,256) into f16 WMMA A-register layout.
// A-frag half index h of lane: Klocal = (h&7) | ((h>>3)<<4) | ((lane>>4)<<3)
// (ISA 7.12.2: lanes 0-15 hold K{0-7,16-23}, lanes 16-31 hold K{8-15,24-31})
// Conv K ordering: K = kh*512 + kw*256 + d_in (tap-major patch concat).
// ---------------------------------------------------------------------------
__global__ void pack_weights_k(const float* __restrict__ conv_w,
                               const float* __restrict__ red_w,
                               _Float16* __restrict__ packedA,
                               _Float16* __restrict__ packedR) {
    int idx = blockIdx.x * blockDim.x + threadIdx.x;
    if (idx < PACKA_ELEMS) {
        int h = idx & 15, lane = (idx >> 4) & 31, kt = (idx >> 9) & 31, mt = idx >> 14;
        int kl = (h & 7) | ((h >> 3) << 4) | ((lane >> 4) << 3);
        int K  = kt * 32 + kl;
        int M  = mt * 16 + (lane & 15);
        int d  = K & 255, kh = K >> 9, kw = (K >> 8) & 1;
        packedA[idx] = (_Float16)conv_w[M * 1024 + d * 4 + kh * 2 + kw];
    } else if (idx < PACKA_ELEMS + PACKR_ELEMS) {
        int i2 = idx - PACKA_ELEMS;
        int h = i2 & 15, lane = (i2 >> 4) & 31, kt = (i2 >> 9) & 7, mt = i2 >> 12;
        int kl = (h & 7) | ((h >> 3) << 4) | ((lane >> 4) << 3);
        int K  = kt * 32 + kl;
        int M  = mt * 16 + (lane & 15);
        packedR[i2] = (_Float16)red_w[M * 256 + K];
    }
}

// Per-thread source pointer for one 32-float K-block of one patch column.
__device__ __forceinline__ const float* tile_src(const float* __restrict__ X,
                                                 const float* __restrict__ P,
                                                 int wg, int ct, int n, int kb) {
    int colL = ct * 16 + n;            // 0..143 within this WG's 4 batches
    int lb   = colL / NPOS;
    int pos  = colL - lb * NPOS;
    int hh   = pos / 6, ww = pos - hh * 6;
    int tap  = kb >> 3;
    int cell = (2 * hh + (tap >> 1)) * GRID_HW + (2 * ww + (tap & 1));
    int d0   = (kb & 7) * 32;
    return (cell < SEQ) ? (X + ((size_t)(wg * BPW + lb) * SEQ + cell) * DIM + d0)
                        : (P + d0);
}

// ---------------------------------------------------------------------------
// Fused: scatter+conv(GEMM,K=1024) -> GELU -> 1x1 reduce(GEMM,K=256) -> GELU
//        -> mean pool -> 16 residual blocks -> readout.  One WG = 4 batches.
// Double-buffered stage-1 B tiles; X read once with non-temporal loads.
// ---------------------------------------------------------------------------
__global__ __launch_bounds__(512)
void fused_readout_k(const float* __restrict__ X,  const float* __restrict__ P,
                     const float* __restrict__ conv_b, const float* __restrict__ red_b,
                     const float* __restrict__ rw1, const float* __restrict__ rb1,
                     const float* __restrict__ rw2, const float* __restrict__ rb2,
                     const float* __restrict__ out_w, const float* __restrict__ out_b,
                     const _Float16* __restrict__ packedA,
                     const _Float16* __restrict__ packedR,
                     float* __restrict__ out) {
    __shared__ _Float16 Bs[2][32 * 32 * 16]; // stage-1 B frags: [buf][kt][lane][16]
    __shared__ _Float16 B2[8 * 32 * 16];     // stage-2 B frags (K=256 channels)
    __shared__ float pooled[BPW * RDIM];     // pooled features -> becomes h
    __shared__ float ybuf[BPW * RDIM];

    const int t    = threadIdx.x;
    const int lane = t & 31;
    const int w    = t >> 5;                 // wave id 0..15; also column-in-tile
    const int wg   = blockIdx.x;
    const int moff = (lane & 16) ? 8 : 0;    // C layout: lanes 16-31 hold M+8

    if (t < BPW * RDIM) pooled[t] = 0.0f;

    // hoist loop-invariant biases
    float biasA[8];
    #pragma unroll
    for (int j = 0; j < 8; ++j) biasA[j] = conv_b[w * 16 + j + moff];
    float biasR[8];
    if (w < 3) {
        #pragma unroll
        for (int j = 0; j < 8; ++j) biasR[j] = red_b[w * 16 + j + moff];
    }

    const v16h* pA = (const v16h*)packedA;
    const v16h* pR = (const v16h*)packedR;

    // ---- prologue: load + store tile 0 into Bs[0] ----
    {
        const f4v* s4 = (const f4v*)tile_src(X, P, wg, 0, w, lane);
        f4v pf[8];
        #pragma unroll
        for (int i = 0; i < 8; ++i) pf[i] = __builtin_nontemporal_load(&s4[i]);
        v16h lo, hi;
        #pragma unroll
        for (int i = 0; i < 4; ++i)
            #pragma unroll
            for (int k = 0; k < 4; ++k) {
                lo[4*i+k] = (_Float16)pf[i + (i & 2)][k];
                hi[4*i+k] = (_Float16)pf[2 + i + (i & 2)][k];
            }
        *(v16h*)&Bs[0][(lane * 32 + w) * 16]      = lo;  // K%32 in 0..7,16..23
        *(v16h*)&Bs[0][(lane * 32 + w + 16) * 16] = hi;  // K%32 in 8..15,24..31
    }

    for (int ct = 0; ct < NCOLT; ++ct) {
        const int cur = ct & 1, nxt = cur ^ 1;

        // ---- issue next tile's global loads (no wait yet) ----
        f4v pf[8];
        if (ct + 1 < NCOLT) {
            const f4v* s4 = (const f4v*)tile_src(X, P, wg, ct + 1, w, lane);
            #pragma unroll
            for (int i = 0; i < 8; ++i) pf[i] = __builtin_nontemporal_load(&s4[i]);
        }

        __syncthreads();   // Bs[cur] ready; previous stage-2 done with B2

        // ---- stage 1: 256xN conv GEMM, K=1024 (32 WMMAs per wave) ----
        v8f c = {0.f, 0.f, 0.f, 0.f, 0.f, 0.f, 0.f, 0.f};
        #pragma unroll 4
        for (int kt = 0; kt < 32; ++kt) {
            v16h a = pA[(w * 32 + kt) * 32 + lane];
            v16h b = *(const v16h*)&Bs[cur][(kt * 32 + lane) * 16];
            c = __builtin_amdgcn_wmma_f32_16x16x32_f16(false, a, false, b,
                                                       (short)0, c, false, false);
        }

        // ---- convert + store next tile into Bs[nxt] ----
        if (ct + 1 < NCOLT) {
            v16h lo, hi;
            #pragma unroll
            for (int i = 0; i < 4; ++i)
                #pragma unroll
                for (int k = 0; k < 4; ++k) {
                    lo[4*i+k] = (_Float16)pf[i + (i & 2)][k];
                    hi[4*i+k] = (_Float16)pf[2 + i + (i & 2)][k];
                }
            *(v16h*)&Bs[nxt][(lane * 32 + w) * 16]      = lo;
            *(v16h*)&Bs[nxt][(lane * 32 + w + 16) * 16] = hi;
        }

        // ---- bias + GELU, repack accumulators into stage-2 B fragments ----
        {
            v8h hp;
            #pragma unroll
            for (int j = 0; j < 8; ++j)
                hp[j] = (_Float16)gelu_fast(c[j] + biasA[j]);
            int lane_b = (lane & 15) | (lane & 16);
            *(v8h*)&B2[((w >> 1) * 32 + lane_b) * 16 + ((w & 1) << 3)] = hp;
        }
        __syncthreads();   // B2 ready

        // ---- stage 2: 48x16 reduce GEMM, K=256 (waves 0..2), GELU, pool ----
        if (w < 3) {
            v8f c2 = {0.f, 0.f, 0.f, 0.f, 0.f, 0.f, 0.f, 0.f};
            #pragma unroll
            for (int kt = 0; kt < 8; ++kt) {
                v16h a = pR[(w * 8 + kt) * 32 + lane];
                v16h b = *(const v16h*)&B2[(kt * 32 + lane) * 16];
                c2 = __builtin_amdgcn_wmma_f32_16x16x32_f16(false, a, false, b,
                                                            (short)0, c2, false, false);
            }
            int colL = ct * 16 + (lane & 15);
            int lb   = colL / NPOS;
            #pragma unroll
            for (int j = 0; j < 8; ++j) {
                float v = gelu_fast(c2[j] + biasR[j]) * (1.0f / 36.0f);
                atomicAdd(&pooled[lb * RDIM + (w * 16 + j + moff)], v); // ds_add_f32
            }
        }
    }
    __syncthreads();

    // ---- residual MLP stack + readout (fp32 VALU, 1 thread per (row,r)) ----
    int row = t / RDIM, r = t - row * RDIM;
    for (int l = 0; l < NLAYER; ++l) {
        if (t < BPW * RDIM) {
            const float* w1 = rw1 + (l * RDIM + r) * RDIM;
            float acc = rb1[l * RDIM + r];
            #pragma unroll 8
            for (int k = 0; k < RDIM; ++k) acc += pooled[row * RDIM + k] * w1[k];
            ybuf[t] = gelu_fast(acc);
        }
        __syncthreads();
        if (t < BPW * RDIM) {
            const float* w2 = rw2 + (l * RDIM + r) * RDIM;
            float acc = rb2[l * RDIM + r];
            #pragma unroll 8
            for (int k = 0; k < RDIM; ++k) acc += ybuf[row * RDIM + k] * w2[k];
            pooled[t] += acc;
        }
        __syncthreads();
    }
    if (t < BPW * RDIM) ybuf[t] = pooled[t] * out_w[r];
    __syncthreads();
    if (t < BPW) {
        float s = out_b[0];
        #pragma unroll 8
        for (int k = 0; k < RDIM; ++k) s += ybuf[t * RDIM + k];
        out[wg * BPW + t] = s;
    }
}

// ---------------------------------------------------------------------------
extern "C" void kernel_launch(void* const* d_in, const int* in_sizes, int n_in,
                              void* d_out, int out_size, void* d_ws, size_t ws_size,
                              hipStream_t stream) {
    const float* X      = (const float*)d_in[0];
    const float* P      = (const float*)d_in[1];
    const float* conv_w = (const float*)d_in[2];
    const float* conv_b = (const float*)d_in[3];
    const float* red_w  = (const float*)d_in[4];
    const float* red_b  = (const float*)d_in[5];
    const float* rw1    = (const float*)d_in[6];
    const float* rb1    = (const float*)d_in[7];
    const float* rw2    = (const float*)d_in[8];
    const float* rb2    = (const float*)d_in[9];
    const float* ow     = (const float*)d_in[10];
    const float* ob     = (const float*)d_in[11];
    float* out = (float*)d_out;

    _Float16* packedA = (_Float16*)d_ws;
    _Float16* packedR = packedA + PACKA_ELEMS;

    const int nPack = PACKA_ELEMS + PACKR_ELEMS;
    pack_weights_k<<<(nPack + 255) / 256, 256, 0, stream>>>(conv_w, red_w, packedA, packedR);
    fused_readout_k<<<BATCH / BPW, 512, 0, stream>>>(X, P, conv_b, red_b,
                                                     rw1, rb1, rw2, rb2, ow, ob,
                                                     packedA, packedR, out);
}
